// CrystalGATEncoder_84129819394655
// MI455X (gfx1250) — matrix operand order repaired and checked by reading, not compiled, not run on hardware
//
#include <hip/hip_runtime.h>

// ---------------------------------------------------------------------------
// CrystalGATEncoder for MI455X (gfx1250, wave32).
// Dense GEMMs (node_proj, GATConv fc, pooling gate MLP) run on
// v_wmma_f32_16x16x32_f16 (f16 inputs, f32 accum) with a 16x64 tile per wave:
// 4 accumulators share one A operand (4x A reuse) and the K-loop is software
// pipelined (next-K loads in flight during current-K WMMAs). Graph
// scatter/gather ops (edge scatter-add, edge softmax, attention aggregation)
// are wave-per-edge f32 atomic kernels — HBM-bandwidth bound at 23.3 TB/s.
// ---------------------------------------------------------------------------

typedef __attribute__((ext_vector_type(16))) _Float16 v16h;
typedef __attribute__((ext_vector_type(8)))  _Float16 v8h;
typedef __attribute__((ext_vector_type(8)))  float    v8f;

#define NN 50000
#define NE 400000
#define NG 64
#define NH 4

static __device__ __forceinline__ void atomicMaxF(float* addr, float val) {
  unsigned int* ua = (unsigned int*)addr;
  unsigned int cur = *ua;
  while (__uint_as_float(cur) < val) {
    unsigned int prev = atomicCAS(ua, cur, __float_as_uint(val));
    if (prev == cur) break;
    cur = prev;
  }
}

// A-operand fragment (16-bit A 16x32 layout): this lane's 16 halfs are
// 16 contiguous halfs (k = 16h + s).
static __device__ __forceinline__ v16h load_a16(const _Float16* ap) {
  v8h a0 = *(const v8h*)(ap);
  v8h a1 = *(const v8h*)(ap + 8);
  return __builtin_shufflevector(a0, a1, 0, 1, 2, 3, 4, 5, 6, 7,
                                 8, 9, 10, 11, 12, 13, 14, 15);
}

// ------------------------- utility kernels ---------------------------------
__global__ __launch_bounds__(256) void k_zero(float* p, size_t n) {
  size_t i = (size_t)blockIdx.x * blockDim.x + threadIdx.x;
  if (i < n) p[i] = 0.0f;
}

__global__ __launch_bounds__(256) void k_fill(float* p, size_t n, float v) {
  size_t i = (size_t)blockIdx.x * blockDim.x + threadIdx.x;
  if (i < n) p[i] = v;
}

// f32 [M,K] -> f16 [M,Kpad] (zero padded)
__global__ __launch_bounds__(256) void k_pack_a(const float* __restrict__ src,
                                                _Float16* __restrict__ dst,
                                                size_t M, int K, int Kpad) {
  size_t i = (size_t)blockIdx.x * blockDim.x + threadIdx.x;
  size_t tot = M * (size_t)Kpad;
  if (i >= tot) return;
  int k = (int)(i % Kpad);
  size_t m = i / Kpad;
  dst[i] = (k < K) ? (_Float16)src[m * (size_t)K + k] : (_Float16)0.0f;
}

// f32 weight [K,N] -> f16 transposed [N,Kpad] (zero padded)
__global__ __launch_bounds__(256) void k_pack_wt(const float* __restrict__ src,
                                                 _Float16* __restrict__ dst,
                                                 int K, int N, int Kpad) {
  size_t i = (size_t)blockIdx.x * blockDim.x + threadIdx.x;
  size_t tot = (size_t)N * Kpad;
  if (i >= tot) return;
  int k = (int)(i % Kpad);
  int n = (int)(i / Kpad);
  dst[i] = (k < K) ? (_Float16)src[(size_t)k * N + n] : (_Float16)0.0f;
}

// ------------------------- WMMA GEMM ---------------------------------------
// C[M,N] = act(A[M,Kpad] @ Bt[N,Kpad]^T + bias). 16x64 output tile per
// wave32: 4 v8f accumulators share one A fragment; K stepped by 32 through
// v_wmma_f32_16x16x32_f16 with next-K operands double-buffered so the loads
// overlap the matrix ops. Requires M%16==0, N%64==0, Kpad%32==0.
__global__ __launch_bounds__(256) void k_wmma_gemm(
    const _Float16* __restrict__ A, const _Float16* __restrict__ Bt,
    const float* __restrict__ bias, float* __restrict__ C,
    int M, int N, int Kpad, int act) {
  int wave = (int)((blockIdx.x * blockDim.x + threadIdx.x) >> 5);
  int lane = threadIdx.x & 31;
  int ntile = N >> 6;                 // 64-wide tiles
  int total = (M >> 4) * ntile;
  if (wave >= total) return;
  int tm = wave / ntile;
  int tn = wave % ntile;
  int hi = lane >> 4;                 // 0: low half, 1: high half
  int l15 = lane & 15;
  const _Float16* arow = A + (size_t)(tm * 16 + l15) * Kpad + hi * 16;
  const _Float16* b0p = Bt + (size_t)(tn * 64 + l15) * Kpad + hi * 16;
  const _Float16* b1p = b0p + (size_t)16 * Kpad;
  const _Float16* b2p = b0p + (size_t)32 * Kpad;
  const _Float16* b3p = b0p + (size_t)48 * Kpad;

  v8f acc0 = {}, acc1 = {}, acc2 = {}, acc3 = {};
  v16h a  = load_a16(arow);
  v16h b0 = *(const v16h*)(b0p);
  v16h b1 = *(const v16h*)(b1p);
  v16h b2 = *(const v16h*)(b2p);
  v16h b3 = *(const v16h*)(b3p);

  for (int k = 32; k < Kpad; k += 32) {
    // next-K operands issued before the current-K WMMAs -> loads overlap math
    v16h an  = load_a16(arow + k);
    v16h bn0 = *(const v16h*)(b0p + k);
    v16h bn1 = *(const v16h*)(b1p + k);
    v16h bn2 = *(const v16h*)(b2p + k);
    v16h bn3 = *(const v16h*)(b3p + k);
    if (k + 32 < Kpad) __builtin_prefetch(arow + k + 32, 0, 2);
    acc0 = __builtin_amdgcn_wmma_f32_16x16x32_f16(false, a, false, b0, (short)0, acc0, false, false);
    acc1 = __builtin_amdgcn_wmma_f32_16x16x32_f16(false, a, false, b1, (short)0, acc1, false, false);
    acc2 = __builtin_amdgcn_wmma_f32_16x16x32_f16(false, a, false, b2, (short)0, acc2, false, false);
    acc3 = __builtin_amdgcn_wmma_f32_16x16x32_f16(false, a, false, b3, (short)0, acc3, false, false);
    a = an; b0 = bn0; b1 = bn1; b2 = bn2; b3 = bn3;
  }
  acc0 = __builtin_amdgcn_wmma_f32_16x16x32_f16(false, a, false, b0, (short)0, acc0, false, false);
  acc1 = __builtin_amdgcn_wmma_f32_16x16x32_f16(false, a, false, b1, (short)0, acc1, false, false);
  acc2 = __builtin_amdgcn_wmma_f32_16x16x32_f16(false, a, false, b2, (short)0, acc2, false, false);
  acc3 = __builtin_amdgcn_wmma_f32_16x16x32_f16(false, a, false, b3, (short)0, acc3, false, false);

  int rbase = tm * 16 + hi * 8;
  v8f accs[4] = {acc0, acc1, acc2, acc3};
#pragma unroll
  for (int t = 0; t < 4; ++t) {
    int colg = tn * 64 + t * 16 + l15;
    float bv = bias ? bias[colg] : 0.0f;
#pragma unroll
    for (int r = 0; r < 8; ++r) {
      float v = accs[t][r] + bv;
      if (act == 1) v = v > 0.0f ? v : 0.0f;
      C[(size_t)(rbase + r) * N + colg] = v;
    }
  }
}

// ------------------------- GNN pieces --------------------------------------
// edge MLP: relu(ef@W1+b1)@W2+b2   [E,3]->[E,3]
__global__ __launch_bounds__(256) void k_edge_mlp(
    const float* __restrict__ ef_in, const float* __restrict__ W1,
    const float* __restrict__ b1, const float* __restrict__ W2,
    const float* __restrict__ b2, float* __restrict__ ef_out, int E) {
  int e = blockIdx.x * blockDim.x + threadIdx.x;
  if (e >= E) return;
  float x0 = ef_in[e * 3 + 0], x1 = ef_in[e * 3 + 1], x2 = ef_in[e * 3 + 2];
  float hdn[6];
#pragma unroll
  for (int j = 0; j < 6; ++j) {
    float v = x0 * W1[0 * 6 + j] + x1 * W1[1 * 6 + j] + x2 * W1[2 * 6 + j] + b1[j];
    hdn[j] = v > 0.0f ? v : 0.0f;
  }
#pragma unroll
  for (int j = 0; j < 3; ++j) {
    float v = b2[j];
#pragma unroll
    for (int i = 0; i < 6; ++i) v += hdn[i] * W2[i * 3 + j];
    ef_out[e * 3 + j] = v;
  }
}

// layer-1 node_proj (4x4) scalar
__global__ __launch_bounds__(256) void k_proj4(
    const float* __restrict__ h, const float* __restrict__ pW,
    const float* __restrict__ pb, float* __restrict__ hp, int n) {
  int i = blockIdx.x * blockDim.x + threadIdx.x;
  if (i >= n * 4) return;
  int r = i >> 2, j = i & 3;
  float v = pb[j];
#pragma unroll
  for (int k = 0; k < 4; ++k) v += h[r * 4 + k] * pW[k * 4 + j];
  hp[i] = v;
}

// temp[src] += [hp[src], ef]; temp[dst] += [hp[dst], ef]   (wave per edge)
__global__ __launch_bounds__(256) void k_scatter(
    const float* __restrict__ hp, const float* __restrict__ ef,
    const int* __restrict__ src, const int* __restrict__ dst,
    float* __restrict__ temp, int E, int infeat, int C) {
  int w = (int)((blockIdx.x * blockDim.x + threadIdx.x) >> 5);
  if (w >= E) return;
  int lane = threadIdx.x & 31;
  int s = src[w], d = dst[w];
  const float* hs = hp + (size_t)s * infeat;
  const float* hd = hp + (size_t)d * infeat;
  float* ts = temp + (size_t)s * C;
  float* td = temp + (size_t)d * C;
  for (int c = lane; c < infeat; c += 32) {
    atomicAdd(&ts[c], hs[c]);
    atomicAdd(&td[c], hd[c]);
  }
  if (lane < 3) {
    float v = ef[(size_t)w * 3 + lane];
    atomicAdd(&ts[infeat + lane], v);
    atomicAdd(&td[infeat + lane], v);
  }
}

// el/er = sum_f z[n,h,f] * attn_{l,r}[h,f]
__global__ __launch_bounds__(256) void k_el_er(
    const float* __restrict__ z, const float* __restrict__ al,
    const float* __restrict__ ar, float* __restrict__ el,
    float* __restrict__ er, int n, int F) {
  int i = blockIdx.x * blockDim.x + threadIdx.x;
  if (i >= n * NH) return;
  int h = i & 3;
  const float* zp = z + (size_t)i * F;
  const float* alp = al + h * F;
  const float* arp = ar + h * F;
  float sl = 0.0f, sr = 0.0f;
  for (int f = 0; f < F; ++f) { float zv = zp[f]; sl += zv * alp[f]; sr += zv * arp[f]; }
  el[i] = sl; er[i] = sr;
}

__global__ __launch_bounds__(256) void k_attn1(
    const float* __restrict__ el, const float* __restrict__ er,
    const int* __restrict__ src, const int* __restrict__ dst,
    float* __restrict__ ev, float* __restrict__ m, int E) {
  int i = blockIdx.x * blockDim.x + threadIdx.x;
  if (i >= E * NH) return;
  int e = i >> 2, h = i & 3;
  float v = el[src[e] * NH + h] + er[dst[e] * NH + h];
  v = v > 0.0f ? v : 0.2f * v;            // leaky_relu 0.2
  ev[i] = v;
  atomicMaxF(&m[dst[e] * NH + h], v);
}

__global__ __launch_bounds__(256) void k_attn2(
    float* __restrict__ ev, const float* __restrict__ m,
    const int* __restrict__ dst, float* __restrict__ sbuf, int E) {
  int i = blockIdx.x * blockDim.x + threadIdx.x;
  if (i >= E * NH) return;
  int e = i >> 2, h = i & 3;
  float ex = __expf(ev[i] - m[dst[e] * NH + h]);
  ev[i] = ex;
  atomicAdd(&sbuf[dst[e] * NH + h], ex);
}

__global__ __launch_bounds__(256) void k_attn3(
    float* __restrict__ ev, const float* __restrict__ sbuf,
    const int* __restrict__ dst, int E) {
  int i = blockIdx.x * blockDim.x + threadIdx.x;
  if (i >= E * NH) return;
  int e = i >> 2, h = i & 3;
  ev[i] = ev[i] / sbuf[dst[e] * NH + h];
}

// rst[dst] += z[src] * a   (wave per edge)
__global__ __launch_bounds__(256) void k_aggregate(
    const float* __restrict__ z, const float* __restrict__ ev,
    const int* __restrict__ src, const int* __restrict__ dst,
    float* __restrict__ rst, int E, int HF, int lf) {
  int w = (int)((blockIdx.x * blockDim.x + threadIdx.x) >> 5);
  if (w >= E) return;
  int lane = threadIdx.x & 31;
  int s = src[w], d = dst[w];
  float a0 = ev[w * NH + 0], a1 = ev[w * NH + 1];
  float a2 = ev[w * NH + 2], a3 = ev[w * NH + 3];
  const float* zs = z + (size_t)s * HF;
  float* rd = rst + (size_t)d * HF;
  for (int c = lane; c < HF; c += 32) {
    int h = c >> lf;
    float am = (h == 0) ? a0 : (h == 1) ? a1 : (h == 2) ? a2 : a3;
    atomicAdd(&rd[c], zs[c] * am);
  }
}

// h_out = elu(rst + gbias)
__global__ __launch_bounds__(256) void k_elu_bias(
    const float* __restrict__ rst, const float* __restrict__ gbias,
    float* __restrict__ out, size_t total, int HF) {
  size_t i = (size_t)blockIdx.x * blockDim.x + threadIdx.x;
  if (i >= total) return;
  int c = (int)(i % HF);
  float x = rst[i] + gbias[c];
  out[i] = x > 0.0f ? x : expm1f(x);
}

// gate[n] = G1[n,:] . gW2 + gb2   (wave per node)
__global__ __launch_bounds__(256) void k_gate(
    const float* __restrict__ G1, const float* __restrict__ gW2,
    const float* __restrict__ gb2, float* __restrict__ gate, int n) {
  int w = (int)((blockIdx.x * blockDim.x + threadIdx.x) >> 5);
  int lane = threadIdx.x & 31;
  if (w >= n) return;
  const float* g = G1 + (size_t)w * 128;
  float p = 0.0f;
  for (int c = lane; c < 128; c += 32) p += g[c] * gW2[c];
#pragma unroll
  for (int o = 16; o; o >>= 1) p += __shfl_xor(p, o, 32);
  if (lane == 0) gate[w] = p + gb2[0];
}

__global__ __launch_bounds__(256) void k_pool_max(
    const float* __restrict__ gate, const int* __restrict__ gid,
    float* __restrict__ gm, int n) {
  int i = blockIdx.x * blockDim.x + threadIdx.x;
  if (i >= n) return;
  atomicMaxF(&gm[gid[i]], gate[i]);
}

__global__ __launch_bounds__(256) void k_pool_sum(
    const float* __restrict__ gate, const int* __restrict__ gid,
    const float* __restrict__ gm, float* __restrict__ gex,
    float* __restrict__ gs, int n) {
  int i = blockIdx.x * blockDim.x + threadIdx.x;
  if (i >= n) return;
  float e = __expf(gate[i] - gm[gid[i]]);
  gex[i] = e;
  atomicAdd(&gs[gid[i]], e);
}

// graph_emb[g] += (gex[n]/gs[g]) * h[n,:]   (wave per node)
__global__ __launch_bounds__(256) void k_pool_emb(
    const float* __restrict__ h, const float* __restrict__ gex,
    const float* __restrict__ gs, const int* __restrict__ gid,
    float* __restrict__ emb, int n) {
  int w = (int)((blockIdx.x * blockDim.x + threadIdx.x) >> 5);
  if (w >= n) return;
  int lane = threadIdx.x & 31;
  int g = gid[w];
  float coef = gex[w] / gs[g];
  const float* hp = h + (size_t)w * 1024;
  float* ep = emb + (size_t)g * 1024;
  for (int c = lane; c < 1024; c += 32) atomicAdd(&ep[c], coef * hp[c]);
}

// mu / clipped logvar heads
__global__ __launch_bounds__(256) void k_heads(
    const float* __restrict__ emb, const float* __restrict__ muW,
    const float* __restrict__ mub, const float* __restrict__ lvW,
    const float* __restrict__ lvb, float* __restrict__ mu,
    float* __restrict__ lv) {
  int i = blockIdx.x * blockDim.x + threadIdx.x;
  if (i >= NG * 64) return;
  int g = i >> 6, j = i & 63;
  const float* e = emb + (size_t)g * 1024;
  float sm = 0.0f, sv = 0.0f;
  for (int k = 0; k < 1024; ++k) {
    float ev = e[k];
    sm += ev * muW[k * 64 + j];
    sv += ev * lvW[k * 64 + j];
  }
  mu[i] = sm + mub[j];
  lv[i] = fminf(10.0f, fmaxf(-10.0f, sv + lvb[j]));
}

// ------------------------- host driver -------------------------------------
static inline unsigned int cdiv(size_t a, size_t b) {
  return (unsigned int)((a + b - 1) / b);
}
static inline int align32i(int x) { return (x + 31) & ~31; }

extern "C" void kernel_launch(void* const* d_in, const int* in_sizes, int n_in,
                              void* d_out, int out_size, void* d_ws, size_t ws_size,
                              hipStream_t stream) {
  const float* node_feat = (const float*)d_in[0];
  const float* edge_feat = (const float*)d_in[1];
  const int* src = (const int*)d_in[2];
  const int* dst = (const int*)d_in[3];
  const int* gids = (const int*)d_in[4];
  const float* P1[10]; const float* P2[10]; const float* P3[10];
  for (int i = 0; i < 10; ++i) P1[i] = (const float*)d_in[5 + i];
  for (int i = 0; i < 10; ++i) P2[i] = (const float*)d_in[15 + i];
  for (int i = 0; i < 10; ++i) P3[i] = (const float*)d_in[25 + i];
  const float* gW1 = (const float*)d_in[35];
  const float* gb1 = (const float*)d_in[36];
  const float* gW2 = (const float*)d_in[37];
  const float* gb2 = (const float*)d_in[38];
  const float* muW = (const float*)d_in[39];
  const float* mub = (const float*)d_in[40];
  const float* lvW = (const float*)d_in[41];
  const float* lvb = (const float*)d_in[42];

  float* out = (float*)d_out;
  float* out_mu = out;               // [64,64]
  float* out_lv = out + 4096;        // [64,64]
  float* hfin   = out + 8192;        // [50000,1024]

  // workspace bump allocator (256B aligned)
  char* base = (char*)d_ws;
  size_t off = 0;
  auto alloc = [&](size_t bytes) -> void* {
    void* p = base + off;
    off = (off + bytes + 255) & ~(size_t)255;
    return p;
  };
  float*    ef    = (float*)alloc((size_t)NE * 3 * 4);
  float*    hp    = (float*)alloc((size_t)NN * 512 * 4);
  float*    temp  = (float*)alloc((size_t)NN * 515 * 4);
  float*    zbuf  = (float*)alloc((size_t)NN * 1024 * 4);
  float*    rst   = (float*)alloc((size_t)NN * 1024 * 4);
  float*    el    = (float*)alloc((size_t)NN * NH * 4);
  float*    er    = (float*)alloc((size_t)NN * NH * 4);
  float*    ev    = (float*)alloc((size_t)NE * NH * 4);
  float*    mbuf  = (float*)alloc((size_t)NN * NH * 4);
  float*    sbuf  = (float*)alloc((size_t)NN * NH * 4);
  _Float16* A16   = (_Float16*)alloc((size_t)NN * 1024 * 2);
  _Float16* B16   = (_Float16*)alloc((size_t)1024 * 1024 * 2);
  float*    h1    = (float*)alloc((size_t)NN * 256 * 4);
  float*    h2    = (float*)alloc((size_t)NN * 512 * 4);
  float*    G1    = (float*)alloc((size_t)NN * 128 * 4);
  float*    gate  = (float*)alloc((size_t)NN * 4);
  float*    gex   = (float*)alloc((size_t)NN * 4);
  float*    gm    = (float*)alloc((size_t)NG * 4);
  float*    gs    = (float*)alloc((size_t)NG * 4);
  float*    gemb  = (float*)alloc((size_t)NG * 1024 * 4);

  auto gemm = [&](const _Float16* A, const _Float16* Bt, const float* bias,
                  float* C, int M, int N, int Kpad, int act) {
    size_t threads = (size_t)(M / 16) * (N / 64) * 32;  // 16x64 tile per wave
    k_wmma_gemm<<<cdiv(threads, 256), 256, 0, stream>>>(A, Bt, bias, C, M, N, Kpad, act);
  };

  auto run_layer = [&](const float* h_in, int infeat, int outfeat,
                       const float* const* P, float* h_out) {
    int C = infeat + 3;
    int Kc = align32i(C);
    int HF = NH * outfeat;
    int lf = (outfeat == 64) ? 6 : (outfeat == 128) ? 7 : 8;
    // edge MLP
    k_edge_mlp<<<cdiv(NE, 256), 256, 0, stream>>>(edge_feat, P[0], P[1], P[2], P[3], ef, NE);
    // node_proj
    if (infeat == 4) {
      k_proj4<<<cdiv((size_t)NN * 4, 256), 256, 0, stream>>>(h_in, P[4], P[5], hp, NN);
    } else {
      k_pack_a<<<cdiv((size_t)NN * infeat, 256), 256, 0, stream>>>(h_in, A16, NN, infeat, infeat);
      k_pack_wt<<<cdiv((size_t)infeat * infeat, 256), 256, 0, stream>>>(P[4], B16, infeat, infeat, infeat);
      gemm(A16, B16, P[5], hp, NN, infeat, infeat, 0);
    }
    // scatter-add into temp
    k_zero<<<cdiv((size_t)NN * C, 256), 256, 0, stream>>>(temp, (size_t)NN * C);
    k_scatter<<<cdiv((size_t)NE * 32, 256), 256, 0, stream>>>(hp, ef, src, dst, temp, NE, infeat, C);
    // z = temp @ fcW  (WMMA)
    k_pack_a<<<cdiv((size_t)NN * Kc, 256), 256, 0, stream>>>(temp, A16, NN, C, Kc);
    k_pack_wt<<<cdiv((size_t)HF * Kc, 256), 256, 0, stream>>>(P[6], B16, C, HF, Kc);
    gemm(A16, B16, nullptr, zbuf, NN, HF, Kc, 0);
    // attention
    k_el_er<<<cdiv((size_t)NN * NH, 256), 256, 0, stream>>>(zbuf, P[7], P[8], el, er, NN, outfeat);
    k_fill<<<cdiv((size_t)NN * NH, 256), 256, 0, stream>>>(mbuf, (size_t)NN * NH, -1e30f);
    k_zero<<<cdiv((size_t)NN * NH, 256), 256, 0, stream>>>(sbuf, (size_t)NN * NH);
    k_attn1<<<cdiv((size_t)NE * NH, 256), 256, 0, stream>>>(el, er, src, dst, ev, mbuf, NE);
    k_attn2<<<cdiv((size_t)NE * NH, 256), 256, 0, stream>>>(ev, mbuf, dst, sbuf, NE);
    k_attn3<<<cdiv((size_t)NE * NH, 256), 256, 0, stream>>>(ev, sbuf, dst, NE);
    // aggregate + bias + elu
    k_zero<<<cdiv((size_t)NN * HF, 256), 256, 0, stream>>>(rst, (size_t)NN * HF);
    k_aggregate<<<cdiv((size_t)NE * 32, 256), 256, 0, stream>>>(zbuf, ev, src, dst, rst, NE, HF, lf);
    k_elu_bias<<<cdiv((size_t)NN * HF, 256), 256, 0, stream>>>(rst, P[9], h_out, (size_t)NN * HF, HF);
  };

  run_layer(node_feat, 4, 64, P1, h1);     // -> [N,256]
  run_layer(h1, 256, 128, P2, h2);         // -> [N,512]
  run_layer(h2, 512, 256, P3, hfin);       // -> [N,1024] (d_out)

  // GlobalAttentionPooling
  k_pack_a<<<cdiv((size_t)NN * 1024, 256), 256, 0, stream>>>(hfin, A16, NN, 1024, 1024);
  k_pack_wt<<<cdiv((size_t)128 * 1024, 256), 256, 0, stream>>>(gW1, B16, 1024, 128, 1024);
  gemm(A16, B16, gb1, G1, NN, 128, 1024, 1);   // relu(h@gW1+gb1)
  k_gate<<<cdiv((size_t)NN * 32, 256), 256, 0, stream>>>(G1, gW2, gb2, gate, NN);
  k_fill<<<1, 256, 0, stream>>>(gm, NG, -1e30f);
  k_zero<<<1, 256, 0, stream>>>(gs, NG);
  k_zero<<<cdiv((size_t)NG * 1024, 256), 256, 0, stream>>>(gemb, (size_t)NG * 1024);
  k_pool_max<<<cdiv(NN, 256), 256, 0, stream>>>(gate, gids, gm, NN);
  k_pool_sum<<<cdiv(NN, 256), 256, 0, stream>>>(gate, gids, gm, gex, gs, NN);
  k_pool_emb<<<cdiv((size_t)NN * 32, 256), 256, 0, stream>>>(hfin, gex, gs, gids, gemb, NN);
  // VAE heads
  k_heads<<<cdiv((size_t)NG * 64, 256), 256, 0, stream>>>(gemb, muW, mub, lvW, lvb, out_mu, out_lv);
}
